// GCN_34110630265430
// MI455X (gfx1250) — matrix operand order repaired and checked by reading, not compiled, not run on hardware
//
#include <hip/hip_runtime.h>
#include <cstdint>

typedef __attribute__((ext_vector_type(16))) _Float16 v16h_t;
typedef __attribute__((ext_vector_type(8)))  _Float16 v8h_t;
typedef __attribute__((ext_vector_type(4)))  _Float16 v4h_t;
typedef __attribute__((ext_vector_type(8)))  float    v8f_t;
typedef __attribute__((ext_vector_type(4)))  float    v4f_t;

#define GCN_NODES 100000
#define GCN_EDGES 3200000
#define GCN_FEAT  256
#define GCN_HID   128
#define GCN_REP   128

// ---------------------------------------------------------------------------
// Pack a [K x 128] f32 weight matrix into WMMA B-fragment order (f16).
// Wave32 B layout for V_WMMA_F32_16X16X32_F16 (CDNA5 ISA):
//   lane l holds column N = l&15 of the 32x16 tile; kh = (l>>4)*8,
//   elems 0..7 -> K = kh+0..7, elems 8..15 -> K = 16+kh+0..7.
// Packed layout: Wp[((t*KSTEPS + s)*32 + lane)*16 + e]
// ---------------------------------------------------------------------------
template<int KDIM>
__global__ __launch_bounds__(256) void gcn_pack_w(const float* __restrict__ W,
                                                  _Float16* __restrict__ Wp) {
    constexpr int KSTEPS = KDIM / 32;
    constexpr int TOTAL  = 8 * KSTEPS * 32 * 16;
    int idx = blockIdx.x * blockDim.x + threadIdx.x;
    if (idx >= TOTAL) return;
    int e    = idx & 15;          // fragment element (selects K)
    int lane = (idx >> 4) & 31;
    int rem  = idx >> 9;
    int s    = rem % KSTEPS;      // K-step (32 per step)
    int t    = rem / KSTEPS;      // N tile (8 tiles of 16 -> 128)
    int k    = s * 32 + ((lane >> 4) << 3) + (e & 7) + ((e >> 3) << 4);
    Wp[idx]  = (_Float16)W[(size_t)k * 128 + t * 16 + (lane & 15)];
}

// ---------------------------------------------------------------------------
// C[M x 128] = A[M x KDIM] * W[KDIM x 128]  via v_wmma_f32_16x16x32_f16.
// Packed weights are staged into LDS once per block (64KB/32KB out of the
// WGP's 320KB); the wmma K-loop is then fed by ds_load_b128 instead of
// global->L0 round-trips. One wave computes a 16-row stripe across all eight
// 16-col tiles (8 x v8f accumulators = 64 VGPRs).
// A-fragment layout (16-bit A 16x32, CDNA5 ISA 7.12.2):
//   lane l: row M = l&15, kh = (l>>4)*8;
//   elems 0..7 -> K = kh+0..7, elems 8..15 -> K = 16+kh+0..7.
// A reads are non-temporal: each activation row is read exactly once, and we
// want the 192MB L2 to stay resident with the scatter buffers instead.
// ---------------------------------------------------------------------------
template<int KDIM, bool AHALF>
__global__ __launch_bounds__(256) void gcn_wmma_gemm(const void* __restrict__ Av,
                                                     const _Float16* __restrict__ Wp,
                                                     float* __restrict__ C, int M) {
    constexpr int KSTEPS = KDIM / 32;
    constexpr int WELEMS = 8 * KSTEPS * 32 * 16;     // packed halves (64KB for K=256)
    __shared__ _Float16 sW[WELEMS];

    // Cooperative stage of packed weights into LDS (16B per thread per step).
    {
        const v8h_t* src = (const v8h_t*)Wp;
        v8h_t*       dst = (v8h_t*)sW;
        constexpr int CHUNKS = WELEMS / 8;
#pragma unroll
        for (int c = 0; c < CHUNKS / 256; ++c)
            dst[c * 256 + threadIdx.x] = src[c * 256 + threadIdx.x];
    }
    __syncthreads();

    const int lane    = threadIdx.x & 31;
    const int wave    = threadIdx.x >> 5;
    const int rowBase = (blockIdx.x * 8 + wave) * 16;
    if (rowBase >= M) return;                 // whole-wave uniform exit; EXEC stays all-ones
    const int mr    = lane & 15;              // A row within tile
    const int khalf = (lane >> 4) << 3;       // 0 or 8

    v8f_t acc[8];
#pragma unroll
    for (int t = 0; t < 8; ++t) acc[t] = (v8f_t)0.0f;

#pragma unroll
    for (int s = 0; s < KSTEPS; ++s) {
        const int kb = s * 32 + khalf;
        v16h_t afrag;
        if constexpr (AHALF) {
            const _Float16* Ah = (const _Float16*)Av + (size_t)(rowBase + mr) * KDIM + kb;
            v8h_t h0 = __builtin_nontemporal_load((const v8h_t*)(Ah));       // K = kb..kb+7
            v8h_t h1 = __builtin_nontemporal_load((const v8h_t*)(Ah + 16));  // K = kb+16..kb+23
            afrag = __builtin_shufflevector(h0, h1, 0,1,2,3,4,5,6,7,8,9,10,11,12,13,14,15);
        } else {
            const float* Af = (const float*)Av + (size_t)(rowBase + mr) * KDIM + kb;
            v4f_t a0 = __builtin_nontemporal_load((const v4f_t*)(Af));
            v4f_t a1 = __builtin_nontemporal_load((const v4f_t*)(Af + 4));
            v4f_t a2 = __builtin_nontemporal_load((const v4f_t*)(Af + 16));
            v4f_t a3 = __builtin_nontemporal_load((const v4f_t*)(Af + 20));
#pragma unroll
            for (int i = 0; i < 4; ++i) {
                afrag[i]      = (_Float16)a0[i];
                afrag[4 + i]  = (_Float16)a1[i];
                afrag[8 + i]  = (_Float16)a2[i];
                afrag[12 + i] = (_Float16)a3[i];
            }
        }
#pragma unroll
        for (int t = 0; t < 8; ++t) {
            v16h_t bfrag = *(const v16h_t*)(sW + ((t * KSTEPS + s) * 32 + lane) * 16);
            acc[t] = __builtin_amdgcn_wmma_f32_16x16x32_f16(
                /*neg_a=*/false, afrag, /*neg_b=*/false, bfrag,
                /*c_mod=*/(short)0, acc[t], /*reuse_a=*/false, /*reuse_b=*/false);
        }
    }

    // D layout: VGPR r, lane l -> M = r + (l>>4)*8, N = l&15
    const int mBase = rowBase + khalf;
    const int nc    = lane & 15;
#pragma unroll
    for (int t = 0; t < 8; ++t) {
#pragma unroll
        for (int r = 0; r < 8; ++r) {
            C[(size_t)(mBase + r) * 128 + t * 16 + nc] = acc[t][r];
        }
    }
}

// ---------------------------------------------------------------------------
// out[n, j] = bias[j]   (seed aggregation buffers / d_out with the bias)
// one float4 per thread
// ---------------------------------------------------------------------------
__global__ __launch_bounds__(256) void gcn_init_bias(float* __restrict__ o,
                                                     const float* __restrict__ b,
                                                     int total4) {
    int i = blockIdx.x * blockDim.x + threadIdx.x;
    if (i < total4) {
        v4f_t bv = *(const v4f_t*)&b[(i & 31) * 4];
        *(v4f_t*)&o[(size_t)i * 4] = bv;
    }
}

// ---------------------------------------------------------------------------
// Edge scatter: one wave32 per edge; lane handles 4 contiguous features.
// agg[dst] += support[src] * w  via hardware global_atomic_add_f32.
// support/agg (51.2 MB each) are L2-resident on MI455X (192 MB L2).
// ---------------------------------------------------------------------------
__global__ __launch_bounds__(256) void gcn_scatter(const float* __restrict__ sup,
                                                   const int* __restrict__ esrc,
                                                   const int* __restrict__ edst,
                                                   const float* __restrict__ ew,
                                                   float* __restrict__ agg, int nE) {
    const int e = (blockIdx.x * blockDim.x + threadIdx.x) >> 5;
    if (e >= nE) return;
    const int lane = threadIdx.x & 31;
    const int s = esrc[e];
    const int d = edst[e];
    const float w = ew[e];
    const v4f_t v = *(const v4f_t*)&sup[(size_t)s * 128 + lane * 4];
    float* o = &agg[(size_t)d * 128 + lane * 4];
    unsafeAtomicAdd(o + 0, v[0] * w);
    unsafeAtomicAdd(o + 1, v[1] * w);
    unsafeAtomicAdd(o + 2, v[2] * w);
    unsafeAtomicAdd(o + 3, v[3] * w);
}

// ---------------------------------------------------------------------------
// h1 = (f16) relu(agg1)    (bias already seeded into agg1)
// one float4 -> half4 per thread; NT load frees L2 lines after last use
// ---------------------------------------------------------------------------
__global__ __launch_bounds__(256) void gcn_relu_cvt(const float* __restrict__ a,
                                                    _Float16* __restrict__ h,
                                                    int total4) {
    int i = blockIdx.x * blockDim.x + threadIdx.x;
    if (i < total4) {
        v4f_t v = __builtin_nontemporal_load((const v4f_t*)&a[(size_t)i * 4]);
        v4h_t r;
#pragma unroll
        for (int j = 0; j < 4; ++j) r[j] = (_Float16)(v[j] > 0.0f ? v[j] : 0.0f);
        *(v4h_t*)&h[(size_t)i * 4] = r;
    }
}

extern "C" void kernel_launch(void* const* d_in, const int* in_sizes, int n_in,
                              void* d_out, int out_size, void* d_ws, size_t ws_size,
                              hipStream_t stream) {
    const float* x    = (const float*)d_in[0];   // [100000 x 256] f32
    const int*   esrc = (const int*)  d_in[1];   // [3.2M]
    const int*   edst = (const int*)  d_in[2];   // [3.2M]
    const float* ew   = (const float*)d_in[3];   // [3.2M]
    const float* W1   = (const float*)d_in[4];   // [256 x 128]
    const float* b1   = (const float*)d_in[5];   // [128]
    const float* W2   = (const float*)d_in[6];   // [128 x 128]
    const float* b2   = (const float*)d_in[7];   // [128]
    float* out        = (float*)d_out;           // [100000 x 128]

    // Workspace layout (≈128.1 MB total):
    char* ws = (char*)d_ws;
    _Float16* Wp1  = (_Float16*)(ws);                         // 64 KB packed W1
    _Float16* Wp2  = (_Float16*)(ws + (64 << 10));            // 32 KB packed W2
    const size_t BUF = (size_t)GCN_NODES * 128 * sizeof(float);   // 51.2 MB
    float*    bufA = (float*)(ws + (128 << 10));              // support1 / support2
    float*    bufB = (float*)(ws + (128 << 10) + BUF);        // agg1
    _Float16* h1   = (_Float16*)(ws + (128 << 10) + 2 * BUF); // 25.6 MB f16 hidden

    const int NF4  = GCN_NODES * 128 / 4;      // 3.2M float4 elements
    const int fblk = NF4 / 256;                // 12500
    const int gemmBlocks = (GCN_NODES / 16 + 7) / 8;   // 6250 waves -> 782 blocks
    const int edgeBlocks = (GCN_EDGES * 32) / 256;     // 400000

    // Layer 1: support1 = X @ W1 (WMMA, f32 A converted inline)
    gcn_pack_w<GCN_FEAT><<<(8 * (GCN_FEAT/32) * 32 * 16 + 255) / 256, 256, 0, stream>>>(W1, Wp1);
    gcn_wmma_gemm<GCN_FEAT, false><<<gemmBlocks, 256, 0, stream>>>(x, Wp1, bufA, GCN_NODES);
    gcn_init_bias<<<fblk, 256, 0, stream>>>(bufB, b1, NF4);
    gcn_scatter<<<edgeBlocks, 256, 0, stream>>>(bufA, esrc, edst, ew, bufB, GCN_EDGES);
    gcn_relu_cvt<<<fblk, 256, 0, stream>>>(bufB, h1, NF4);

    // Layer 2: support2 = h1 @ W2 (WMMA, f16 A direct); aggregate into d_out
    gcn_pack_w<GCN_HID><<<(8 * (GCN_HID/32) * 32 * 16 + 255) / 256, 256, 0, stream>>>(W2, Wp2);
    gcn_wmma_gemm<GCN_HID, true><<<gemmBlocks, 256, 0, stream>>>(h1, Wp2, bufA, GCN_NODES);
    gcn_init_bias<<<fblk, 256, 0, stream>>>(out, b2, NF4);
    gcn_scatter<<<edgeBlocks, 256, 0, stream>>>(bufA, esrc, edst, ew, out, GCN_EDGES);
}
